// AgentAttention_87411174408775
// MI455X (gfx1250) — matrix operand order repaired and checked by reading, not compiled, not run on hardware
//
#include <hip/hip_runtime.h>
#include <hip/hip_bf16.h>
#include <math.h>

// ---------------- problem constants (fixed by reference) ----------------
#define BB    16            // batch
#define HH    56            // height
#define WW    56            // width
#define NN    3136          // tokens = 56*56
#define CC    256           // channels
#define NH    8             // heads
#define HD    32            // head dim
#define AG    49            // agents (7*7)
#define TT    1568          // top-k = NN/2
#define MM    (BB*NN)       // 50176 GEMM rows
#define SCALE 0.17677669529663687f   // 32^-0.5

typedef __bf16 v16bf __attribute__((ext_vector_type(16)));
typedef float  v8f   __attribute__((ext_vector_type(8)));

__device__ __forceinline__ unsigned short f2bf(float f) {
    unsigned int u = __float_as_uint(f);
    u += 0x7FFFu + ((u >> 16) & 1u);   // round-to-nearest-even
    return (unsigned short)(u >> 16);
}

// ---------------- f32 -> bf16 convert ----------------
__global__ void k_cvt_bf16(const float* __restrict__ src,
                           unsigned short* __restrict__ dst, int n) {
    int i = blockIdx.x * blockDim.x + threadIdx.x;
    if (i < n) dst[i] = f2bf(src[i]);
}

// ---------------- bf16 WMMA GEMM:  C[M,N] = A[M,K] * B[N,K]^T (+bias) ----
// One block per 16-row M tile: A tile async-staged to LDS once, A fragments
// held in registers; per N-group all 16 B loads issue first (one clause),
// then the 8-WMMA chain drains the load counter progressively.
__global__ void k_gemm_bf16(const unsigned short* __restrict__ A,
                            const unsigned short* __restrict__ B,
                            float* __restrict__ C,
                            const float* __restrict__ bias,
                            int M, int N, int ldc) {
    __shared__ unsigned short As[16 * 256];       // 8 KB A tile (16 x K)

    const int ntiles  = N >> 4;
    const int ngroups = ntiles >> 3;              // N-groups of 8 wave-tiles
    const int mt   = blockIdx.x;
    const int wave = threadIdx.x >> 5;
    const int tid  = threadIdx.x;

    // ---- cooperative async copy: contiguous 8KB A tile -> LDS ----
    {
        unsigned lds = ((unsigned)(unsigned long long)(void*)As) + tid * 32u;
        unsigned long long gsrc =
            (unsigned long long)(A + (size_t)mt * 16 * 256) + (unsigned)tid * 32u;
        asm volatile(
            "global_load_async_to_lds_b128 %0, %1, off\n\t"
            "global_load_async_to_lds_b128 %0, %1, off offset:16"
            :: "v"(lds), "v"(gsrc) : "memory");
        asm volatile("s_wait_asynccnt 0x0" ::: "memory");
    }
    __syncthreads();

    const int lane = threadIdx.x & 31;
    const int r    = lane & 15;       // A-row / B-col within tile
    const int half = lane >> 4;

    // ---- load all 8 A fragments (full K=256) into registers once ----
    union frag { v16bf v; uint4 q[2]; };
    frag a8[8];
    const unsigned short* Ar = As + r * 256;      // LDS row
#pragma unroll
    for (int i = 0; i < 8; ++i) {
        a8[i].q[0] = *(const uint4*)(Ar + i * 32 + half * 8);
        a8[i].q[1] = *(const uint4*)(Ar + i * 32 + 16 + half * 8);
    }

    // ---- loop over N-groups: batch-load B, then 8 WMMAs ----
    for (int g = 0; g < ngroups; ++g) {
        const int nt = g * 8 + wave;
        const unsigned short* Br = B + (size_t)(nt * 16 + r) * 256;
        frag b8[8];
#pragma unroll
        for (int i = 0; i < 8; ++i) {
            b8[i].q[0] = *(const uint4*)(Br + i * 32 + half * 8);
            b8[i].q[1] = *(const uint4*)(Br + i * 32 + 16 + half * 8);
        }
        v8f acc = {};
#pragma unroll
        for (int i = 0; i < 8; ++i)
            acc = __builtin_amdgcn_wmma_f32_16x16x32_bf16(
                      false, a8[i].v, false, b8[i].v, (short)0, acc, false, false);

        const int   col  = nt * 16 + r;
        const float bv   = bias ? bias[col] : 0.0f;
        const int   row0 = mt * 16 + half * 8;
#pragma unroll
        for (int i = 0; i < 8; ++i)
            C[(size_t)(row0 + i) * ldc + col] = acc[i] + bv;
    }
}

// ---------------- agent pooling: 8x8 mean of q_lin -> agent[b,h,a,d] ------
__global__ void k_agent_pool(const float* __restrict__ qkv,
                             float* __restrict__ agent) {
    int idx = blockIdx.x * blockDim.x + threadIdx.x;
    if (idx >= BB * AG * CC) return;
    int b  = idx / (AG * CC);
    int a  = (idx / CC) % AG;
    int cc = idx % CC;
    int p = a / 7, q = a % 7;
    float s = 0.f;
    for (int dy = 0; dy < 8; ++dy)
        for (int dx = 0; dx < 8; ++dx) {
            int m = (p * 8 + dy) * WW + (q * 8 + dx);
            s += qkv[(size_t)(b * NN + m) * 768 + cc];
        }
    int h = cc >> 5, d = cc & 31;
    agent[(size_t)((b * NH + h) * AG + a) * HD + d] = s * (1.0f / 64.0f);
}

// ---------------- per-head token mean of q ----------------
__global__ void k_token_mean(const float* __restrict__ qkv,
                             float* __restrict__ tmean) {
    int idx = blockIdx.x * blockDim.x + threadIdx.x;
    if (idx >= BB * NH * NN) return;
    int b = idx / (NH * NN);
    int h = (idx / NN) % NH;
    int m = idx % NN;
    const float* qp = qkv + (size_t)(b * NN + m) * 768 + h * HD;
    float s = 0.f;
#pragma unroll
    for (int d = 0; d < HD; ++d) s += qp[d];
    tmean[(size_t)(b * NH + h) * NN + m] = s * (1.0f / HD);
}

// ---------------- exact top-half sorted indices via bitonic sort ----------
__device__ __forceinline__ unsigned int sortable(float f) {
    unsigned int u = __float_as_uint(f);
    return (u & 0x80000000u) ? ~u : (u | 0x80000000u);
}

__global__ void k_topk(const float* __restrict__ tmean,
                       int* __restrict__ top_idx) {
    __shared__ unsigned long long keys[4096];
    int bh  = blockIdx.x;           // (b*8+h)
    int tid = threadIdx.x;
    for (int i = tid; i < 4096; i += 256) {
        unsigned long long kk = 0ull;
        if (i < NN) {
            unsigned int u = sortable(tmean[(size_t)bh * NN + i]);
            kk = ((unsigned long long)u << 32) | (unsigned)(0xFFFFFFFFu - i);
        }
        keys[i] = kk;
    }
    __syncthreads();
    for (int k = 2; k <= 4096; k <<= 1)
        for (int j = k >> 1; j > 0; j >>= 1) {
            for (int i = tid; i < 4096; i += 256) {
                int ixj = i ^ j;
                if (ixj > i) {
                    unsigned long long x = keys[i], y = keys[ixj];
                    bool up = ((i & k) == 0);        // descending sort
                    if ((x < y) == up) { keys[i] = y; keys[ixj] = x; }
                }
            }
            __syncthreads();
        }
    for (int t = tid; t < TT; t += 256)
        top_idx[(size_t)bh * TT + t] =
            (int)(0xFFFFFFFFu - (unsigned)(keys[t] & 0xFFFFFFFFu));
}

// ---------------- bias pipeline (7x7 -> 56x56 bilerp + axis biases) -------
__global__ void k_bias(const float* __restrict__ an_bias,
                       const float* __restrict__ na_bias,
                       const float* __restrict__ ah_bias,
                       const float* __restrict__ aw_bias,
                       const float* __restrict__ ha_bias,
                       const float* __restrict__ wa_bias,
                       float* __restrict__ pos_bias,   // [h, 49, 3136]
                       float* __restrict__ agt_bias) { // [h, 3136, 49]
    int idx = blockIdx.x * blockDim.x + threadIdx.x;
    if (idx >= NH * AG * NN) return;
    int h = idx / (AG * NN);
    int a = (idx / NN) % AG;
    int m = idx % NN;
    int y = m / WW, x = m % WW;

    float sy = (y + 0.5f) * 0.125f - 0.5f;
    float sx = (x + 0.5f) * 0.125f - 0.5f;
    int y0 = (int)floorf(sy), x0 = (int)floorf(sx);
    float fy = sy - y0, fx = sx - x0;
    int y0c = min(6, max(0, y0)),     x0c = min(6, max(0, x0));
    int y1c = min(6, max(0, y0 + 1)), x1c = min(6, max(0, x0 + 1));

    const float* an = an_bias + (size_t)(h * AG + a) * 49;
    const float* na = na_bias + (size_t)(h * AG + a) * 49;
    float ban = (1 - fy) * ((1 - fx) * an[y0c * 7 + x0c] + fx * an[y0c * 7 + x1c]) +
                fy       * ((1 - fx) * an[y1c * 7 + x0c] + fx * an[y1c * 7 + x1c]);
    float bna = (1 - fy) * ((1 - fx) * na[y0c * 7 + x0c] + fx * na[y0c * 7 + x1c]) +
                fy       * ((1 - fx) * na[y1c * 7 + x0c] + fx * na[y1c * 7 + x1c]);

    pos_bias[(size_t)(h * AG + a) * NN + m] =
        ban + ah_bias[(size_t)(h * AG + a) * HH + y]
            + aw_bias[(size_t)(h * AG + a) * WW + x];
    agt_bias[((size_t)h * NN + m) * AG + a] =
        bna + ha_bias[(size_t)(h * HH + y) * AG + a]
            + wa_bias[(size_t)(h * WW + x) * AG + a];
}

// ---------------- agent attention: softmax_n(agent*scale . k + pb) @ v ----
__global__ void k_agent_attn(const float* __restrict__ qkv,
                             const float* __restrict__ agent,
                             const float* __restrict__ pos_bias,
                             float* __restrict__ agent_v) {
    __shared__ float arow[HD];
    __shared__ float sbuf[NN];
    __shared__ float red[256];
    __shared__ float red2[256 * HD];

    int blk = blockIdx.x;            // ((b*8+h)*49 + a)
    int a   = blk % AG;
    int bh  = blk / AG;
    int h   = bh % NH;
    int b   = bh / NH;
    int tid = threadIdx.x;

    if (tid < HD) arow[tid] = agent[((size_t)bh * AG + a) * HD + tid] * SCALE;
    __syncthreads();

    const float* pb = pos_bias + (size_t)(h * AG + a) * NN;
    float lmax = -1e30f;
    for (int m = tid; m < NN; m += 256) {
        const float* kp = qkv + (size_t)(b * NN + m) * 768 + 256 + h * HD;
        float s = 0.f;
#pragma unroll
        for (int d = 0; d < HD; ++d) s += arow[d] * kp[d];
        s += pb[m];
        sbuf[m] = s;
        lmax = fmaxf(lmax, s);
    }
    red[tid] = lmax; __syncthreads();
    for (int off = 128; off > 0; off >>= 1) {
        if (tid < off) red[tid] = fmaxf(red[tid], red[tid + off]);
        __syncthreads();
    }
    float mx = red[0]; __syncthreads();

    float lsum = 0.f;
    for (int m = tid; m < NN; m += 256) {
        float p = expf(sbuf[m] - mx);
        sbuf[m] = p;
        lsum += p;
    }
    red[tid] = lsum; __syncthreads();
    for (int off = 128; off > 0; off >>= 1) {
        if (tid < off) red[tid] += red[tid + off];
        __syncthreads();
    }
    float denom = red[0]; __syncthreads();

    float acc[HD];
#pragma unroll
    for (int d = 0; d < HD; ++d) acc[d] = 0.f;
    for (int m = tid; m < NN; m += 256) {
        float p = sbuf[m];
        const float* vp = qkv + (size_t)(b * NN + m) * 768 + 512 + h * HD;
#pragma unroll
        for (int d = 0; d < HD; ++d) acc[d] += p * vp[d];
    }
#pragma unroll
    for (int d = 0; d < HD; ++d) red2[tid * HD + d] = acc[d];
    __syncthreads();
    if (tid < HD) {
        float s = 0.f;
        for (int i = 0; i < 256; ++i) s += red2[i * HD + tid];
        agent_v[((size_t)bh * AG + a) * HD + tid] = s / denom;
    }
}

// --- fused: gather top rows, row-interp (resize), softmax_49, @agent_v ----
__global__ void k_attn2(const float* __restrict__ qkv,
                        const float* __restrict__ agent,
                        const float* __restrict__ agent_v,
                        const int* __restrict__ top_idx,
                        const float* __restrict__ agt_bias,
                        float* __restrict__ out_attn) {
    __shared__ float sag[AG * HD];
    __shared__ float sav[AG * HD];
    int nchunk = (NN + 255) / 256;                  // 13
    int bh = blockIdx.x / nchunk;
    int m  = (blockIdx.x % nchunk) * 256 + threadIdx.x;
    int h  = bh % NH;
    int b  = bh / NH;

    for (int i = threadIdx.x; i < AG * HD; i += 256) {
        sag[i] = agent[(size_t)bh * AG * HD + i];
        sav[i] = agent_v[(size_t)bh * AG * HD + i];
    }
    __syncthreads();
    if (m >= NN) return;

    // bilinear resize 1568 -> 3136 along the rank axis
    float t = 0.5f * m - 0.25f;
    int   j = (int)floorf(t);
    float f = t - (float)j;
    if (j < 0)        { j = 0; f = 0.f; }
    if (j > TT - 1)   { j = TT - 1; f = 0.f; }
    int j1 = min(j + 1, TT - 1);
    int r0 = top_idx[(size_t)bh * TT + j];
    int r1 = top_idx[(size_t)bh * TT + j1];

    const float* q0 = qkv + (size_t)(b * NN + r0) * 768 + h * HD;
    const float* q1 = qkv + (size_t)(b * NN + r1) * 768 + h * HD;
    float qi[HD];
#pragma unroll
    for (int d = 0; d < HD; ++d)
        qi[d] = ((1.f - f) * q0[d] + f * q1[d]) * SCALE;

    const float* ab = agt_bias + ((size_t)h * NN + m) * AG;
    float s[AG];
    float mx = -1e30f;
    for (int a = 0; a < AG; ++a) {
        float acc = ab[a];
#pragma unroll
        for (int d = 0; d < HD; ++d) acc += qi[d] * sag[a * HD + d];
        s[a] = acc;
        mx = fmaxf(mx, acc);
    }
    float sum = 0.f;
    for (int a = 0; a < AG; ++a) { s[a] = expf(s[a] - mx); sum += s[a]; }
    float inv = 1.f / sum;

    float* op = out_attn + (size_t)(b * NN + m) * CC + h * HD;
#pragma unroll
    for (int d = 0; d < HD; ++d) {
        float acc = 0.f;
        for (int a = 0; a < AG; ++a) acc += s[a] * sav[a * HD + d];
        op[d] = acc * inv;
    }
}

// ---------------- depthwise 3x3 on v + add + bf16 convert -----------------
__global__ void k_dwconv_add(const float* __restrict__ qkv,
                             const float* __restrict__ dwc_w,
                             const float* __restrict__ dwc_b,
                             const float* __restrict__ out_attn,
                             unsigned short* __restrict__ out_bf) {
    int idx = blockIdx.x * blockDim.x + threadIdx.x;
    if (idx >= MM * CC) return;
    int cc = idx % CC;
    int m  = (idx / CC) % NN;
    int b  = idx / (CC * NN);
    int y = m / WW, x = m % WW;

    float acc = dwc_b[cc];
    const float* wp = dwc_w + cc * 9;
#pragma unroll
    for (int dy = -1; dy <= 1; ++dy) {
        int yy = y + dy;
        if (yy < 0 || yy >= HH) continue;
#pragma unroll
        for (int dx = -1; dx <= 1; ++dx) {
            int xx = x + dx;
            if (xx < 0 || xx >= WW) continue;
            acc += qkv[(size_t)(b * NN + yy * WW + xx) * 768 + 512 + cc] *
                   wp[(dy + 1) * 3 + (dx + 1)];
        }
    }
    out_bf[idx] = f2bf(out_attn[idx] + acc);
}

// ------------------------------ launcher ----------------------------------
extern "C" void kernel_launch(void* const* d_in, const int* in_sizes, int n_in,
                              void* d_out, int out_size, void* d_ws, size_t ws_size,
                              hipStream_t stream) {
    (void)in_sizes; (void)n_in; (void)out_size; (void)ws_size;

    const float* x       = (const float*)d_in[0];
    const float* Wq      = (const float*)d_in[3];
    const float* Wkv     = (const float*)d_in[4];
    const float* Wproj   = (const float*)d_in[5];
    const float* bproj   = (const float*)d_in[6];
    const float* dwc_w   = (const float*)d_in[7];
    const float* dwc_b   = (const float*)d_in[8];
    const float* an_bias = (const float*)d_in[9];
    const float* na_bias = (const float*)d_in[10];
    const float* ah_bias = (const float*)d_in[11];
    const float* aw_bias = (const float*)d_in[12];
    const float* ha_bias = (const float*)d_in[13];
    const float* wa_bias = (const float*)d_in[14];
    float* out = (float*)d_out;

    // workspace carve-up (256B aligned)
    char* p = (char*)d_ws;
    auto carve = [&](size_t bytes) {
        void* r = (void*)p;
        p += (bytes + 255) & ~(size_t)255;
        return r;
    };
    unsigned short* xbf   = (unsigned short*)carve((size_t)MM * CC * 2);
    unsigned short* Wbf   = (unsigned short*)carve((size_t)768 * CC * 2);
    unsigned short* Wpbf  = (unsigned short*)carve((size_t)CC * CC * 2);
    float* qkv      = (float*)carve((size_t)MM * 768 * 4);
    float* agent    = (float*)carve((size_t)BB * NH * AG * HD * 4);
    float* tmean    = (float*)carve((size_t)BB * NH * NN * 4);
    int*   top_idx  = (int*)carve((size_t)BB * NH * TT * 4);
    float* pos_bias = (float*)carve((size_t)NH * AG * NN * 4);
    float* agt_bias = (float*)carve((size_t)NH * NN * AG * 4);
    float* agent_v  = (float*)carve((size_t)BB * NH * AG * HD * 4);
    float* out_attn = (float*)carve((size_t)MM * CC * 4);
    unsigned short* outbf = (unsigned short*)carve((size_t)MM * CC * 2);

    // 1. bf16 conversions
    {
        int n = MM * CC;
        k_cvt_bf16<<<(n + 255) / 256, 256, 0, stream>>>(x, xbf, n);
        k_cvt_bf16<<<(CC * CC + 255) / 256, 256, 0, stream>>>(Wq, Wbf, CC * CC);
        k_cvt_bf16<<<(2 * CC * CC + 255) / 256, 256, 0, stream>>>(Wkv, Wbf + CC * CC, 2 * CC * CC);
        k_cvt_bf16<<<(CC * CC + 255) / 256, 256, 0, stream>>>(Wproj, Wpbf, CC * CC);
    }
    // 2. QKV GEMM: 50176 x 768 x 256 (wmma bf16, async A-tile staging)
    k_gemm_bf16<<<MM / 16, 256, 0, stream>>>(xbf, Wbf, qkv, nullptr,
                                             MM, 768, 768);
    // 3. agent pooling + token means
    k_agent_pool<<<(BB * AG * CC + 255) / 256, 256, 0, stream>>>(qkv, agent);
    k_token_mean<<<(BB * NH * NN + 255) / 256, 256, 0, stream>>>(qkv, tmean);
    // 4. sorted top-half indices
    k_topk<<<BB * NH, 256, 0, stream>>>(tmean, top_idx);
    // 5. bias pipelines
    k_bias<<<(NH * AG * NN + 255) / 256, 256, 0, stream>>>(
        an_bias, na_bias, ah_bias, aw_bias, ha_bias, wa_bias, pos_bias, agt_bias);
    // 6. agent attention -> agent_v
    k_agent_attn<<<BB * NH * AG, 256, 0, stream>>>(qkv, agent, pos_bias, agent_v);
    // 7. fused second attention -> out_attn
    {
        int nchunk = (NN + 255) / 256;
        k_attn2<<<BB * NH * nchunk, 256, 0, stream>>>(qkv, agent, agent_v,
                                                      top_idx, agt_bias, out_attn);
    }
    // 8. depthwise conv + add + bf16
    k_dwconv_add<<<(MM * CC + 255) / 256, 256, 0, stream>>>(
        qkv, dwc_w, dwc_b, out_attn, outbf);
    // 9. projection GEMM with bias -> d_out
    k_gemm_bf16<<<MM / 16, 256, 0, stream>>>(outbf, Wpbf, out, bproj,
                                             MM, CC, CC);
}
